// Decoder_60017872995168
// MI455X (gfx1250) — matrix operand order repaired and checked by reading, not compiled, not run on hardware
//
#include <hip/hip_runtime.h>
#include <math.h>

// ---------------------------------------------------------------------------
// Types for CDNA5 WMMA (wave32): v16bf A/B fragments, v8f accumulator.
// ---------------------------------------------------------------------------
typedef __attribute__((ext_vector_type(16))) __bf16 v16bf;
typedef __attribute__((ext_vector_type(8)))  float  v8f;
typedef __attribute__((ext_vector_type(4)))  int    v4i;

// async global->LDS pointer types (per clang diagnostic: int4 in AS1 / AS3)
typedef __attribute__((address_space(1))) v4i gas_v4i;
typedef __attribute__((address_space(3))) v4i las_v4i;

union BFrag {
  v16bf v;
  uint4 q[2];
  __bf16 h[16];
  unsigned short s[16];
};

union Pack4 {
  __bf16 h[4];
  ushort4 u;
};

// native f32 -> bf16 (RNE) via hardware convert
__device__ __forceinline__ unsigned short f2bf(float f) {
  union { __bf16 h; unsigned short u; } c;
  c.h = (__bf16)f;
  return c.u;
}

// Async global->LDS path (gfx1250): guard keeps both toolchains compiling.
#if defined(__AMDGCN__) && __has_builtin(__builtin_amdgcn_global_load_async_to_lds_b128)
#define USE_ASYNC_LDS 1
#else
#define USE_ASYNC_LDS 0
#endif

// ---------------------------------------------------------------------------
// Problem constants
// ---------------------------------------------------------------------------
#define BB   128
#define SS   1024
#define HH   1024
#define EE   512
#define MTOT (BB * SS)          // 131072 rows of the big GEMM

// ---------------------------------------------------------------------------
// Workspace layout (bytes)
// ---------------------------------------------------------------------------
static constexpr size_t PK_W1  = 0;
static constexpr size_t SZ_HxH = 1024ull * 1024 * 2;          // bf16
static constexpr size_t PK_W2  = PK_W1 + SZ_HxH;
static constexpr size_t PK_K0  = PK_W2 + SZ_HxH;
static constexpr size_t SZ_K0  = 1536ull * 3072 * 2;
static constexpr size_t PK_K1  = PK_K0 + SZ_K0;
static constexpr size_t SZ_KL  = 1024ull * 3072 * 2;
static constexpr size_t PK_RK1 = PK_K1 + SZ_KL;
static constexpr size_t PK_K2  = PK_RK1 + SZ_KL;
static constexpr size_t PK_RK2 = PK_K2 + SZ_KL;
static constexpr size_t OFF_Q     = PK_RK2 + SZ_KL;
static constexpr size_t OFF_SCORE = OFF_Q     + 128ull * 1024 * 4;
static constexpr size_t OFF_CTXP  = OFF_SCORE + 131072ull * 4;
static constexpr size_t OFF_XT    = OFF_CTXP  + 8ull * 128 * 1024 * 4;
static constexpr size_t OFF_MX    = OFF_XT    + 128ull * 1536 * 4;
static constexpr size_t OFF_MI    = OFF_MX    + 128ull * 3072 * 4;
static constexpr size_t OFF_H0    = OFF_MI    + 128ull * 3072 * 4;
static constexpr size_t OFF_Y0    = OFF_H0    + 131072ull * 4;
static constexpr size_t OFF_H1    = OFF_Y0    + 131072ull * 4;
static constexpr size_t OFF_Y1    = OFF_H1    + 131072ull * 4;   // total ~48 MB

// ---------------------------------------------------------------------------
// Pack f32 weight [K][N] into bf16 WMMA-B fragment tiles:
// pk[((nt*KT + kt)*32 + lane)*16 + h] = W[kt*32 + (h/8)*16 + (lane/16)*8 + h%8][nt*16 + lane%16]
// so each lane's 16 halfs are its B-fragment contiguous in memory.
// ---------------------------------------------------------------------------
__global__ __launch_bounds__(256)
void pack_bf16_kernel(const float* __restrict__ W, unsigned short* __restrict__ pk,
                      int K, int N) {
  const size_t e = (size_t)blockIdx.x * 256 + threadIdx.x;
  if (e >= (size_t)K * N) return;
  const int k = (int)(e / N), n = (int)(e % N);
  const int kt = k >> 5, kl = k & 31;
  const int lane = (n & 15) | ((kl & 8) << 1);
  const int h = (kl & 7) | ((kl >> 1) & 8);
  const int nt = n >> 4;
  const int ktc = K >> 5;
  pk[(((size_t)nt * ktc + kt) * 32 + lane) * 16 + h] = f2bf(W[e]);
}

__global__ __launch_bounds__(256)
void init_score_kernel(float* __restrict__ score, const float* __restrict__ vb) {
  const int i = blockIdx.x * 256 + threadIdx.x;
  if (i < MTOT) score[i] = vb[0];
}

// ---------------------------------------------------------------------------
// Generic WMMA GEMM: C[M,N] = A[M,K] @ Bpk + bias.  A f32 (converted inline),
// Bpk pre-packed bf16 fragments.  One 16x16 tile per wave, grid-strided.
// ---------------------------------------------------------------------------
__global__ __launch_bounds__(256)
void gemm_bf16_kernel(const float* __restrict__ A, const unsigned short* __restrict__ Bpk,
                      const float* __restrict__ bias, float* __restrict__ C,
                      int M, int K, int N) {
  const int wave = threadIdx.x >> 5;
  const int lane = threadIdx.x & 31;
  const int lr = lane & 15, hf = lane >> 4;
  const int mtc = M >> 4, ntc = N >> 4, ktc = K >> 5;
  const int tiles = mtc * ntc;
  for (int t = blockIdx.x * 8 + wave; t < tiles; t += gridDim.x * 8) {
    const int mt = t % mtc, nt = t / mtc;
    v8f acc = {0, 0, 0, 0, 0, 0, 0, 0};
    const float* arow = A + (size_t)(mt * 16 + lr) * K + hf * 8;
    const unsigned short* bp = Bpk + ((size_t)nt * ktc * 32 + lane) * 16;
    for (int kt = 0; kt < ktc; ++kt) {
      const float* ap = arow + kt * 32;
      const float4 f0 = *(const float4*)(ap);
      const float4 f1 = *(const float4*)(ap + 4);
      const float4 f2 = *(const float4*)(ap + 16);
      const float4 f3 = *(const float4*)(ap + 20);
      BFrag a;
      a.h[0]  = (__bf16)f0.x; a.h[1]  = (__bf16)f0.y; a.h[2]  = (__bf16)f0.z; a.h[3]  = (__bf16)f0.w;
      a.h[4]  = (__bf16)f1.x; a.h[5]  = (__bf16)f1.y; a.h[6]  = (__bf16)f1.z; a.h[7]  = (__bf16)f1.w;
      a.h[8]  = (__bf16)f2.x; a.h[9]  = (__bf16)f2.y; a.h[10] = (__bf16)f2.z; a.h[11] = (__bf16)f2.w;
      a.h[12] = (__bf16)f3.x; a.h[13] = (__bf16)f3.y; a.h[14] = (__bf16)f3.z; a.h[15] = (__bf16)f3.w;
      BFrag b;
      const uint4* bq = (const uint4*)(bp + (size_t)kt * 512);
      b.q[0] = bq[0]; b.q[1] = bq[1];
      acc = __builtin_amdgcn_wmma_f32_16x16x32_bf16(false, a.v, false, b.v,
                                                    (short)0, acc, false, false);
    }
    const int n = nt * 16 + lr;
    const float bn = bias ? bias[n] : 0.f;
    float* cp = C + n;
#pragma unroll
    for (int i = 0; i < 8; ++i)
      cp[(size_t)(mt * 16 + hf * 8 + i) * N] = acc[i] + bn;
  }
}

// ---------------------------------------------------------------------------
// Fused attention-score kernel:
//   score[m] += sum_n tanh( (enc @ w1)[m,n] + b1[n] + q[b,n] ) * vw[n]
// 64 rows per WG, 512 threads = 16 waves = 4 row-tiles x 4 n-groups x 16 tiles.
// A (enc) staged f32->bf16 into LDS; packed w1 tiles staged into LDS once per
// k-chunk via async global->LDS (ASYNCcnt) and shared by all waves.
// ---------------------------------------------------------------------------
#define ATT_A_BYTES (64 * 40 * 2)           // [64 rows][40 halfs] padded
#define ATT_B_BYTES (64 * 32 * 24 * 2)      // [64 nt][32 lanes][24 halfs] padded
#define ATT_SMEM    (ATT_A_BYTES + ATT_B_BYTES)

__global__ __launch_bounds__(512)
void attn_fused_kernel(const float* __restrict__ enc, const float* __restrict__ q,
                       const float* __restrict__ b1, const float* __restrict__ vw,
                       const unsigned short* __restrict__ pkW1,
                       float* __restrict__ score) {
  extern __shared__ char smem[];
  unsigned short* Ash = (unsigned short*)smem;
  unsigned short* Bsh = (unsigned short*)(smem + ATT_A_BYTES);

  const int tid  = threadIdx.x;
  const int wave = tid >> 5;
  const int lane = tid & 31;
  const int lr   = lane & 15;
  const int hf   = lane >> 4;
  const int rt   = wave & 3;    // row tile 0..3
  const int ng   = wave >> 2;   // n-group 0..3
  const int wg   = blockIdx.x;
  const size_t row0 = (size_t)wg * 64;
  const int b = wg >> 4;        // 64-row block never crosses a batch

  const int ar = tid >> 3;              // 0..63  (A stage row)
  const int ac = (tid & 7) << 2;        // 0..28  (A stage col*4)
  const float* asrc = enc + (row0 + ar) * 1024 + ac;
  const unsigned short* afp = Ash + (rt * 16 + lr) * 40 + hf * 8;

  v8f acc[16];
  const v8f vzero = {0, 0, 0, 0, 0, 0, 0, 0};
#pragma unroll
  for (int t = 0; t < 16; ++t) acc[t] = vzero;

  for (int kt = 0; kt < 32; ++kt) {
    __syncthreads();
    // stage A chunk: 64 rows x 32 k, f32 -> bf16 (native cvt), one b64 store
    {
      const float4 av = *(const float4*)(asrc + kt * 32);
      Pack4 p;
      p.h[0] = (__bf16)av.x; p.h[1] = (__bf16)av.y;
      p.h[2] = (__bf16)av.z; p.h[3] = (__bf16)av.w;
      *(ushort4*)(Ash + ar * 40 + ac) = p.u;
      if (kt + 1 < 32) __builtin_prefetch(asrc + (kt + 1) * 32, 0, 1);
    }
    // stage B chunk: 64 nt x 32 lanes x 32B fragments (48B padded stride)
#pragma unroll
    for (int jj = 0; jj < 4; ++jj) {
      const int slot = tid + jj * 512;
      const int nt = slot >> 5, bl = slot & 31;
      const unsigned short* gsrc = pkW1 + (((size_t)nt * 32 + kt) * 32 + bl) * 16;
      unsigned short* ldst = Bsh + (size_t)(nt * 32 + bl) * 24;
#if USE_ASYNC_LDS
      // INST_OFFSET applies to both LDS and global addresses (ISA 15.18.3)
      __builtin_amdgcn_global_load_async_to_lds_b128(
          (gas_v4i*)gsrc, (las_v4i*)ldst, 0, 0);
      __builtin_amdgcn_global_load_async_to_lds_b128(
          (gas_v4i*)gsrc, (las_v4i*)ldst, 16, 0);
#else
      const uint4* gq = (const uint4*)gsrc;
      uint4* dq = (uint4*)ldst;
      dq[0] = gq[0]; dq[1] = gq[1];
#endif
    }
#if USE_ASYNC_LDS
#if __has_builtin(__builtin_amdgcn_s_wait_asynccnt)
    __builtin_amdgcn_s_wait_asynccnt(0);
#else
    asm volatile("s_wait_asynccnt 0x0" ::: "memory");
#endif
#endif
    __syncthreads();

    BFrag a;
    a.q[0] = *(const uint4*)(afp);
    a.q[1] = *(const uint4*)(afp + 16);
#pragma unroll
    for (int t = 0; t < 16; ++t) {
      const int nt = ng * 16 + t;
      BFrag bf;
      const uint4* bp = (const uint4*)(Bsh + (size_t)(nt * 32 + lane) * 24);
      bf.q[0] = bp[0]; bf.q[1] = bp[1];
      acc[t] = __builtin_amdgcn_wmma_f32_16x16x32_bf16(false, a.v, false, bf.v,
                                                       (short)0, acc[t], false, false);
    }
  }

  // fused epilogue: tanh + dot with vw, reduced entirely in registers
  float sp[8];
#pragma unroll
  for (int i = 0; i < 8; ++i) sp[i] = 0.f;
  const float* qrow = q + (size_t)b * 1024;
#pragma unroll
  for (int t = 0; t < 16; ++t) {
    const int n = (ng * 16 + t) * 16 + lr;
    const float add = b1[n] + qrow[n];
    const float vwn = vw[n];
#pragma unroll
    for (int i = 0; i < 8; ++i)
      sp[i] += tanhf(acc[t][i] + add) * vwn;
  }
#pragma unroll
  for (int i = 0; i < 8; ++i) {
    float s = sp[i];
    for (int off = 1; off < 16; off <<= 1) s += __shfl_xor(s, off, 32);
    if (lr == 0)
      atomicAdd(&score[row0 + rt * 16 + hf * 8 + i], s);
  }
}

// ---------------------------------------------------------------------------
// Softmax over S per batch (exps kept in registers; two LDS tree reductions)
// ---------------------------------------------------------------------------
__global__ __launch_bounds__(256)
void softmax_kernel(const float* __restrict__ score, float* __restrict__ attn) {
  __shared__ float red[256];
  const int b = blockIdx.x, tid = threadIdx.x;
  const float* s = score + (size_t)b * 1024;
  float v[4];
#pragma unroll
  for (int p = 0; p < 4; ++p) v[p] = s[tid + p * 256];
  float m = fmaxf(fmaxf(v[0], v[1]), fmaxf(v[2], v[3]));
  red[tid] = m; __syncthreads();
  for (int st = 128; st > 0; st >>= 1) {
    if (tid < st) red[tid] = fmaxf(red[tid], red[tid + st]);
    __syncthreads();
  }
  m = red[0]; __syncthreads();
  float ls = 0.f;
#pragma unroll
  for (int p = 0; p < 4; ++p) { v[p] = expf(v[p] - m); ls += v[p]; }
  red[tid] = ls; __syncthreads();
  for (int st = 128; st > 0; st >>= 1) {
    if (tid < st) red[tid] += red[tid + st];
    __syncthreads();
  }
  const float inv = 1.f / red[0];
  float* o = attn + (size_t)b * 1024;
#pragma unroll
  for (int p = 0; p < 4; ++p) o[tid + p * 256] = v[p] * inv;
}

// ---------------------------------------------------------------------------
// Context partials: ctxp[sc][b][h] = sum_{s in chunk sc} attn[b,s]*enc[b,s,h]
// ---------------------------------------------------------------------------
__global__ __launch_bounds__(256)
void context_part_kernel(const float* __restrict__ enc, const float* __restrict__ attn,
                         float* __restrict__ ctxp) {
  const int b = blockIdx.x >> 3, sc = blockIdx.x & 7;
  const int h4 = threadIdx.x << 2;
  float4 acc = {0, 0, 0, 0};
  for (int s = sc * 128; s < sc * 128 + 128; ++s) {
    const float w = attn[(size_t)b * 1024 + s];
    const float4 e = *(const float4*)(enc + ((size_t)b * 1024 + s) * 1024 + h4);
    acc.x += w * e.x; acc.y += w * e.y; acc.z += w * e.z; acc.w += w * e.w;
  }
  *(float4*)(ctxp + ((size_t)sc * 128 + b) * 1024 + h4) = acc;
}

// xt = concat(context, emb[x])  -> [128, 1536]
__global__ __launch_bounds__(256)
void build_xt_kernel(const float* __restrict__ ctxp, const int* __restrict__ x,
                     const float* __restrict__ emb, float* __restrict__ xt) {
  const int idx = blockIdx.x * 256 + threadIdx.x;
  if (idx >= 128 * 1536) return;
  const int b = idx / 1536, c = idx % 1536;
  float v;
  if (c < 1024) {
    v = 0.f;
    for (int sc = 0; sc < 8; ++sc) v += ctxp[((size_t)sc * 128 + b) * 1024 + c];
  } else {
    v = emb[(size_t)x[b] * EE + (c - 1024)];
  }
  xt[(size_t)b * 1536 + c] = v;
}

// ---------------------------------------------------------------------------
// GRU gates. mi==nullptr -> recurrent input is just bias_mi (h_prev == 0 case
// still passes hprev==nullptr).  mx/mi already include their bias rows.
// ---------------------------------------------------------------------------
__global__ __launch_bounds__(256)
void gru_gates_kernel(const float* __restrict__ mx, const float* __restrict__ mi,
                      const float* __restrict__ bias_mi, const float* __restrict__ hprev,
                      float* __restrict__ hout, int act_relu) {
  const int idx = blockIdx.x * 256 + threadIdx.x;
  if (idx >= 131072) return;
  const int bb = idx >> 10, j = idx & 1023;
  const size_t o = (size_t)bb * 3072 + j;
  const float xz = mx[o], xr = mx[o + 1024], xh = mx[o + 2048];
  float iz, ir, ih;
  if (mi) { iz = mi[o]; ir = mi[o + 1024]; ih = mi[o + 2048]; }
  else    { iz = bias_mi[j]; ir = bias_mi[j + 1024]; ih = bias_mi[j + 2048]; }
  const float z = 1.f / (1.f + expf(-(xz + iz)));
  const float r = 1.f / (1.f + expf(-(xr + ir)));
  float hh = xh + r * ih;
  if (act_relu) hh = fmaxf(hh, 0.f);
  const float hp = hprev ? hprev[idx] : 0.f;
  hout[idx] = z * hp + (1.f - z) * hh;
}

// BatchNorm (training stats, biased variance, eps = 1e-3), per column.
__global__ __launch_bounds__(256)
void bn_kernel(const float* __restrict__ hin, const float* __restrict__ gamma,
               const float* __restrict__ beta, float* __restrict__ y) {
  __shared__ float ssum[8][32];
  __shared__ float ssq[8][32];
  const int c = threadIdx.x & 31;
  const int g = threadIdx.x >> 5;
  const int col = blockIdx.x * 32 + c;
  float s = 0.f, q2 = 0.f;
  for (int bb = g * 16; bb < g * 16 + 16; ++bb) {
    const float v = hin[(size_t)bb * 1024 + col];
    s += v; q2 += v * v;
  }
  ssum[g][c] = s; ssq[g][c] = q2;
  __syncthreads();
  if (g == 0) {
    float S = 0.f, Q = 0.f;
    for (int k = 0; k < 8; ++k) { S += ssum[k][c]; Q += ssq[k][c]; }
    ssum[0][c] = S; ssq[0][c] = Q;
  }
  __syncthreads();
  const float mu  = ssum[0][c] * (1.f / 128.f);
  const float var = ssq[0][c] * (1.f / 128.f) - mu * mu;
  const float inv = rsqrtf(var + 0.001f);
  const float gm = gamma[col], bt = beta[col];
  for (int bb = g * 16; bb < g * 16 + 16; ++bb) {
    const size_t o = (size_t)bb * 1024 + col;
    y[o] = gm * (hin[o] - mu) * inv + bt;
  }
}

// ---------------------------------------------------------------------------
extern "C" void kernel_launch(void* const* d_in, const int* in_sizes, int n_in,
                              void* d_out, int out_size, void* d_ws, size_t ws_size,
                              hipStream_t stream) {
  (void)in_sizes; (void)n_in; (void)out_size; (void)ws_size;
  const int*   x     = (const int*)d_in[0];
  const float* hid   = (const float*)d_in[1];
  const float* enc   = (const float*)d_in[2];
  const float* w1    = (const float*)d_in[3];
  const float* b1    = (const float*)d_in[4];
  const float* w2    = (const float*)d_in[5];
  const float* b2    = (const float*)d_in[6];
  const float* vw    = (const float*)d_in[7];
  const float* vb    = (const float*)d_in[8];
  const float* emb   = (const float*)d_in[9];
  const float* k0    = (const float*)d_in[10];
  const float* bias0 = (const float*)d_in[12];
  const float* gam0  = (const float*)d_in[13];
  const float* bet0  = (const float*)d_in[14];
  const float* k1    = (const float*)d_in[15];
  const float* rk1   = (const float*)d_in[16];
  const float* bias1 = (const float*)d_in[17];
  const float* gam1  = (const float*)d_in[18];
  const float* bet1  = (const float*)d_in[19];
  const float* k2    = (const float*)d_in[20];
  const float* rk2   = (const float*)d_in[21];
  const float* bias2 = (const float*)d_in[22];
  const float* gam2  = (const float*)d_in[23];
  const float* bet2  = (const float*)d_in[24];

  char* ws = (char*)d_ws;
  unsigned short* pkW1  = (unsigned short*)(ws + PK_W1);
  unsigned short* pkW2  = (unsigned short*)(ws + PK_W2);
  unsigned short* pkK0  = (unsigned short*)(ws + PK_K0);
  unsigned short* pkK1  = (unsigned short*)(ws + PK_K1);
  unsigned short* pkRK1 = (unsigned short*)(ws + PK_RK1);
  unsigned short* pkK2  = (unsigned short*)(ws + PK_K2);
  unsigned short* pkRK2 = (unsigned short*)(ws + PK_RK2);
  float* Q     = (float*)(ws + OFF_Q);
  float* score = (float*)(ws + OFF_SCORE);
  float* ctxp  = (float*)(ws + OFF_CTXP);
  float* xtb   = (float*)(ws + OFF_XT);
  float* MX    = (float*)(ws + OFF_MX);
  float* MI    = (float*)(ws + OFF_MI);
  float* H0    = (float*)(ws + OFF_H0);
  float* Y0    = (float*)(ws + OFF_Y0);
  float* H1    = (float*)(ws + OFF_H1);
  float* Y1    = (float*)(ws + OFF_Y1);

  float* out  = (float*)d_out;
  float* y2   = out;                // layer_output
  float* h2   = out + 131072;       // h2
  float* attn = out + 262144;       // attn_w

  // --- pack weights to bf16 WMMA-B fragment layout (one-time per call) ---
  pack_bf16_kernel<<<4096,  256, 0, stream>>>(w1,  pkW1,  1024, 1024);
  pack_bf16_kernel<<<4096,  256, 0, stream>>>(w2,  pkW2,  1024, 1024);
  pack_bf16_kernel<<<18432, 256, 0, stream>>>(k0,  pkK0,  1536, 3072);
  pack_bf16_kernel<<<12288, 256, 0, stream>>>(k1,  pkK1,  1024, 3072);
  pack_bf16_kernel<<<12288, 256, 0, stream>>>(rk1, pkRK1, 1024, 3072);
  pack_bf16_kernel<<<12288, 256, 0, stream>>>(k2,  pkK2,  1024, 3072);
  pack_bf16_kernel<<<12288, 256, 0, stream>>>(rk2, pkRK2, 1024, 3072);

  // --- attention ---
  init_score_kernel<<<512, 256, 0, stream>>>(score, vb);
  gemm_bf16_kernel<<<64, 256, 0, stream>>>(hid, pkW2, b2, Q, 128, 1024, 1024);
  attn_fused_kernel<<<MTOT / 64, 512, ATT_SMEM, stream>>>(enc, Q, b1, vw, pkW1, score);
  softmax_kernel<<<128, 256, 0, stream>>>(score, attn);
  context_part_kernel<<<1024, 256, 0, stream>>>(enc, attn, ctxp);
  build_xt_kernel<<<768, 256, 0, stream>>>(ctxp, x, emb, xtb);

  // --- GRU layer 0 (h_prev == 0 -> recurrent matmul degenerates to bias) ---
  gemm_bf16_kernel<<<192, 256, 0, stream>>>(xtb, pkK0, bias0, MX, 128, 1536, 3072);
  gru_gates_kernel<<<512, 256, 0, stream>>>(MX, nullptr, bias0 + 3072, nullptr, H0, 1);
  bn_kernel<<<32, 256, 0, stream>>>(H0, gam0, bet0, Y0);

  // --- GRU layer 1 ---
  gemm_bf16_kernel<<<192, 256, 0, stream>>>(Y0, pkK1, bias1,        MX, 128, 1024, 3072);
  gemm_bf16_kernel<<<192, 256, 0, stream>>>(H0, pkRK1, bias1 + 3072, MI, 128, 1024, 3072);
  gru_gates_kernel<<<512, 256, 0, stream>>>(MX, MI, nullptr, H0, H1, 1);
  bn_kernel<<<32, 256, 0, stream>>>(H1, gam1, bet1, Y1);

  // --- GRU layer 2 (identity activation; h2 and y2 go straight to d_out) ---
  gemm_bf16_kernel<<<192, 256, 0, stream>>>(Y1, pkK2, bias2,        MX, 128, 1024, 3072);
  gemm_bf16_kernel<<<192, 256, 0, stream>>>(H1, pkRK2, bias2 + 3072, MI, 128, 1024, 3072);
  gru_gates_kernel<<<512, 256, 0, stream>>>(MX, MI, nullptr, H1, h2, 0);
  bn_kernel<<<32, 256, 0, stream>>>(h2, gam2, bet2, y2);
}
